// LeWinTransformerBlock_59184649339032
// MI455X (gfx1250) — compile-verified, matrix operand order checked by reading, not run on hardware
//
#include <hip/hip_runtime.h>
#include <hip/hip_bf16.h>

// LeWin transformer block for MI455X (gfx1250), bf16 WMMA + fp32 accumulate,
// TDM (tensor_load_to_lds) weight staging in the attention kernel.
//
// Workspace layout (bytes), total ~134.4 MB:
//   [0)                 xattn  fp32  B*HW*C            = 134217728
//   [134217728)         wqkvT  bf16  192*64            = 24576
//   [+24576)            wpT    bf16  64*64             = 8192
//   [+8192)             w1T    bf16  256*64            = 32768
//   [+32768)            w2T    bf16  64*256            = 32768
//   [+32768)            biasT  fp32  4*64*64           = 65536

typedef __bf16 bf16;
typedef __attribute__((ext_vector_type(8)))  bf16  bf16x8;
typedef __attribute__((ext_vector_type(16))) bf16  bf16x16;
typedef __attribute__((ext_vector_type(8)))  float f32x8;
typedef __attribute__((ext_vector_type(4)))  unsigned int u32x4;
typedef __attribute__((ext_vector_type(8)))  int i32x8;
typedef __attribute__((ext_vector_type(4)))  int i32x4;

#define NB    8
#define CCH   64
#define WIN   8
#define SHIFT 4
#define NTOK  64   // WIN*WIN
#define HID   256

#define OFF_WQKVT 134217728ull
#define OFF_WPT   (OFF_WQKVT + 24576ull)
#define OFF_W1T   (OFF_WPT + 8192ull)
#define OFF_W2T   (OFF_W1T + 32768ull)
#define OFF_BIAS  (OFF_W2T + 32768ull)

// ---- TDM: 1-D bulk copy global -> LDS (ISA ch.7/8, D# groups per §8.3/8.4) -
// data_size = 8B elements; tile_dim0 = n8 (<=65535); 2-D+ groups zeroed.
__device__ __forceinline__ void tdm_load_1d(unsigned lds_off, const void* gptr,
                                            unsigned nbytes) {
  unsigned long long ga = (unsigned long long)gptr;
  unsigned n8 = nbytes >> 3;
  u32x4 g0;
  g0[0] = 1u;                                             // count=1 (valid D#)
  g0[1] = lds_off;                                        // lds_addr [63:32]
  g0[2] = (unsigned)(ga & 0xFFFFFFFFull);                 // global_addr[31:0]
  g0[3] = (unsigned)((ga >> 32) & 0x1FFFFFFull) | (2u << 30); // [56:32]|type=2
  i32x8 g1 = {};
  g1[0] = (3 << 16);                                      // data_size = 8B
  g1[1] = (int)((n8 & 0xFFFFu) << 16);                    // tensor_dim0[15:0]
  g1[2] = (int)((n8 >> 16) & 0xFFFFu);                    // tensor_dim0[31:16]
  g1[3] = (int)(n8 << 16);                                // tile_dim0
  g1[4] = 0;                                              // tile_dim1/2 unused
  g1[5] = (int)n8;                                        // dim0_stride[31:0]
  g1[6] = 0; g1[7] = 0;
  i32x4 z4 = {};
#if defined(__clang_major__) && (__clang_major__ >= 23)
  i32x8 z8 = {};
  __builtin_amdgcn_tensor_load_to_lds(g0, g1, z4, z4, z8, 0);
#else
  __builtin_amdgcn_tensor_load_to_lds(g0, g1, z4, z4, 0);
#endif
}

// ---- WMMA fragment helpers (CDNA5 ISA 7.12.2 layouts, wave32) --------------
// A (16x32 bf16): lane l holds row m=l&15; elements 0..7 = K 8g..8g+7,
// elements 8..15 = K 8g+16..8g+23, g=l>>4.  B (32x16) identical with the row
// index replaced by output column n (load from transposed n-major buffer).
__device__ __forceinline__ bf16x16 load_frag(const bf16* src, int rowBase,
                                             int kBase, int ld, int lane) {
  int r = rowBase + (lane & 15);
  int g = (lane >> 4) & 1;
  const bf16* p = src + r * ld + kBase + 8 * g;
  bf16x8 lo = *(const bf16x8*)(p);
  bf16x8 hi = *(const bf16x8*)(p + 16);
  return __builtin_shufflevector(lo, hi, 0,1,2,3,4,5,6,7,8,9,10,11,12,13,14,15);
}

// C/D (16x16 f32, v8f): lane l, vgpr r -> m = r + 8*(l>>4), n = l&15.
__device__ __forceinline__ void store_frag_rm(bf16* dst, const f32x8 acc,
                                              int mBase, int nBase, int ld, int lane) {
  int n  = nBase + (lane & 15);
  int m0 = mBase + 8 * ((lane >> 4) & 1);
#pragma unroll
  for (int r = 0; r < 8; ++r) dst[(m0 + r) * ld + n] = (bf16)acc[r];
}

__device__ __forceinline__ void store_frag_tr(bf16* dst, const f32x8 acc,
                                              int nBase, int mBase, int ld, int lane) {
  int n  = nBase + (lane & 15);
  int m0 = mBase + 8 * ((lane >> 4) & 1);
  bf16x8 v;
#pragma unroll
  for (int r = 0; r < 8; ++r) v[r] = (bf16)acc[r];
  *(bf16x8*)(dst + n * ld + m0) = v;   // contiguous 16B ds_store
}

#define WMMA_BF16(a, b, c) \
  __builtin_amdgcn_wmma_f32_16x16x32_bf16(false, (a), false, (b), (short)0, (c), false, false)

__device__ __forceinline__ float gelu_exact(float u) {
  return 0.5f * u * (1.0f + erff(u * 0.70710678118f));
}

// ---- K0: weight convert/transpose + relative-position bias table -----------
__global__ __launch_bounds__(256) void prep_kernel(
    const float* __restrict__ wq, const float* __restrict__ wkv,
    const float* __restrict__ wp, const float* __restrict__ w1,
    const float* __restrict__ w2, const float* __restrict__ rpb,
    bf16* wqkvT, bf16* wpT, bf16* w1T, bf16* w2T, float* biasT) {
  int i = blockIdx.x * 256 + threadIdx.x;
  if (i < 12288) {                       // q|k|v transposed: rows = out channel
    int n = i / 64, k = i % 64;
    float v = (n < 64) ? wq[k * 64 + n] : wkv[k * 128 + (n - 64)];
    wqkvT[n * 64 + k] = (bf16)v;
  } else if (i < 16384) {
    int j = i - 12288; int n = j / 64, k = j % 64;
    wpT[n * 64 + k] = (bf16)wp[k * 64 + n];
  } else if (i < 32768) {
    int j = i - 16384; int n = j / 64, k = j % 64;
    w1T[n * 64 + k] = (bf16)w1[k * HID + n];
  } else if (i < 49152) {
    int j = i - 32768; int n = j / 256, k = j % 256;
    w2T[n * 256 + k] = (bf16)w2[k * 64 + n];
  } else if (i < 65536) {                // biasT[h][i][j]
    int j = i - 49152;
    int h = j >> 12, ij = j & 4095, ii = ij >> 6, jj = ij & 63;
    int ri = ii >> 3, ci = ii & 7, rj = jj >> 3, cj = jj & 7;
    int idx = (ri - rj + WIN - 1) * (2 * WIN - 1) + (ci - cj + WIN - 1);
    biasT[j] = rpb[idx * 4 + h];
  }
}

// ---- K1: fused LN1 + shifted-window attention + proj + residual ------------
// LDS carve (dynamic, 114688 B):
//   [0)       xln   [64][64]  bf16   8192
//   [8192)    qbuf  [4][64][32] bf16 16384 (K padded to 32, zeros)
//   [24576)   kbuf  [4][64][32] bf16 16384
//   [40960)   vT    [4][16][64] bf16 8192
//   [49152)   S     [64][64]  f32   16384
//   [65536)   P     [64][64]  bf16  8192
//   [73728)   obuf  [64][64]  bf16  8192
//   [81920)   lwqkv [192][64] bf16  24576   <- TDM staged
//   [106496)  lwp   [64][64]  bf16  8192    <- TDM staged
#define SM_LWQKV 81920u
#define SM_LWP   106496u
__global__ __launch_bounds__(128) void attn_kernel(
    const float* __restrict__ x, const float* __restrict__ ln1g,
    const float* __restrict__ ln1b, const float* __restrict__ bq,
    const float* __restrict__ bkv, const float* __restrict__ bp,
    const bf16* __restrict__ wqkvT, const bf16* __restrict__ wpT,
    const float* __restrict__ biasT, float* __restrict__ xattn) {
  extern __shared__ char smem[];
  bf16*  xln  = (bf16*)smem;
  bf16*  qbuf = xln + 64 * 64;
  bf16*  kbuf = qbuf + 4 * 64 * 32;
  bf16*  vT   = kbuf + 4 * 64 * 32;
  float* S    = (float*)(vT + 4 * 16 * 64);
  bf16*  P    = (bf16*)(S + 64 * 64);
  bf16*  obuf = P + 64 * 64;
  bf16*  lwqkv = (bf16*)(smem + SM_LWQKV);
  bf16*  lwp   = (bf16*)(smem + SM_LWP);

  const int tid = threadIdx.x, lane = tid & 31, wid = tid >> 5;
  const int w = blockIdx.x;
  const int b = w >> 10, wi = w & 1023, wr = wi >> 5, wc = wi & 31;

  // Kick off the weight DMA first (wave 0 issues; overlaps LN below).
  if (wid == 0) {
    tdm_load_1d(SM_LWQKV, wqkvT, 192 * 64 * 2);
    tdm_load_1d(SM_LWP,   wpT,   64 * 64 * 2);
  }
  __builtin_prefetch(biasT, 0, 3);

  // zero the K-pad region of q/k (qbuf,kbuf contiguous: 16384 bf16 = 8192 int)
  {
    int* z = (int*)qbuf;
    for (int i = tid; i < 8192; i += 128) z[i] = 0;
  }

  // LayerNorm + roll(-SHIFT) gather: one thread per token
  if (tid < NTOK) {
    int r = tid >> 3, c = tid & 7;
    int h0 = (wr * WIN + r + SHIFT) & 255;
    int w0 = (wc * WIN + c + SHIFT) & 255;
    const float* px = x + ((size_t)(b * 65536 + h0 * 256 + w0)) * CCH;
    float v[CCH]; float m = 0.f;
#pragma unroll
    for (int j = 0; j < CCH; ++j) { v[j] = px[j]; m += v[j]; }
    m *= (1.f / CCH);
    float var = 0.f;
#pragma unroll
    for (int j = 0; j < CCH; ++j) { float d = v[j] - m; var += d * d; }
    float rs = rsqrtf(var * (1.f / CCH) + 1e-5f);
#pragma unroll
    for (int j = 0; j < CCH; ++j)
      xln[tid * 64 + j] = (bf16)((v[j] - m) * rs * ln1g[j] + ln1b[j]);
  }
  if (wid == 0) __builtin_amdgcn_s_wait_tensorcnt(0);  // weight DMA done
  __syncthreads();

  // QKV GEMM: M=64, K=64, Nout=192 -> 4x12 tiles, K-loop of 2
  const float scale = 0.25f;  // (C/HEADS)^-0.5
  for (int idx = wid; idx < 48; idx += 4) {
    int mt = idx / 12, nt = idx % 12;
    f32x8 acc = {};
    for (int ks = 0; ks < 64; ks += 32) {
      bf16x16 a  = load_frag(xln,   16 * mt, ks, 64, lane);
      bf16x16 bm = load_frag(lwqkv, 16 * nt, ks, 64, lane);
      acc = WMMA_BF16(a, bm, acc);
    }
    int n = lane & 15;
    if (nt < 4) {                       // Q head nt (pre-scaled)
      float bias = bq[nt * 16 + n];
#pragma unroll
      for (int r = 0; r < 8; ++r) acc[r] = (acc[r] + bias) * scale;
      store_frag_rm(qbuf + nt * 64 * 32, acc, 16 * mt, 0, 32, lane);
    } else if (nt < 8) {                // K head nt-4
      float bias = bkv[(nt - 4) * 16 + n];
#pragma unroll
      for (int r = 0; r < 8; ++r) acc[r] += bias;
      store_frag_rm(kbuf + (nt - 4) * 64 * 32, acc, 16 * mt, 0, 32, lane);
    } else {                            // V head nt-8, store transposed
      float bias = bkv[64 + (nt - 8) * 16 + n];
#pragma unroll
      for (int r = 0; r < 8; ++r) acc[r] += bias;
      store_frag_tr(vT + (nt - 8) * 16 * 64, acc, 0, 16 * mt, 64, lane);
    }
  }
  __syncthreads();

  const int gr = (wr == 31), gc = (wc == 31);  // shift-mask bands exist?
  for (int h = 0; h < 4; ++h) {
    // scores: wave wid owns M-tile wid; K=16 (zero-padded to 32)
    {
      int mt = wid;
      for (int nt = 0; nt < 4; ++nt) {
        f32x8 acc = {};
        bf16x16 a  = load_frag(qbuf + h * 64 * 32, 16 * mt, 0, 32, lane);
        bf16x16 bm = load_frag(kbuf + h * 64 * 32, 16 * nt, 0, 32, lane);
        acc = WMMA_BF16(a, bm, acc);
        int n = lane & 15, j = 16 * nt + n;
        int rj = j >> 3, cj = j & 7;
        int gj = (gr ? (rj < 4 ? 1 : 2) : 0) * 3 + (gc ? (cj < 4 ? 1 : 2) : 0);
#pragma unroll
        for (int r = 0; r < 8; ++r) {
          int i2 = 16 * mt + 8 * ((lane >> 4) & 1) + r;
          int ri = i2 >> 3, ci = i2 & 7;
          int gi = (gr ? (ri < 4 ? 1 : 2) : 0) * 3 + (gc ? (ci < 4 ? 1 : 2) : 0);
          float v = acc[r] + biasT[h * 4096 + i2 * 64 + j] +
                    ((gi != gj) ? -100.f : 0.f);
          S[i2 * 64 + j] = v;
        }
      }
    }
    __syncthreads();
    // softmax over rows (one thread per row)
    if (tid < NTOK) {
      float mx = -1e30f;
      for (int j = 0; j < 64; ++j) mx = fmaxf(mx, S[tid * 64 + j]);
      float e[64]; float sum = 0.f;
#pragma unroll
      for (int j = 0; j < 64; ++j) { e[j] = __expf(S[tid * 64 + j] - mx); sum += e[j]; }
      float inv = 1.f / sum;
#pragma unroll
      for (int j = 0; j < 64; ++j) P[tid * 64 + j] = (bf16)(e[j] * inv);
    }
    __syncthreads();
    // P @ V: M=64, K=64, N=16
    {
      int mt = wid;
      f32x8 acc = {};
      for (int ks = 0; ks < 64; ks += 32) {
        bf16x16 a  = load_frag(P, 16 * mt, ks, 64, lane);
        bf16x16 bm = load_frag(vT + h * 16 * 64, 0, ks, 64, lane);
        acc = WMMA_BF16(a, bm, acc);
      }
      store_frag_rm(obuf, acc, 16 * mt, h * 16, 64, lane);
    }
    __syncthreads();
  }

  // output projection + roll(+SHIFT) scatter + residual
  for (int idx = wid; idx < 16; idx += 4) {
    int mt = idx >> 2, nt = idx & 3;
    f32x8 acc = {};
    for (int ks = 0; ks < 64; ks += 32) {
      bf16x16 a  = load_frag(obuf, 16 * mt, ks, 64, lane);
      bf16x16 bm = load_frag(lwp,  16 * nt, ks, 64, lane);
      acc = WMMA_BF16(a, bm, acc);
    }
    int n = lane & 15, c = 16 * nt + n;
    float bias = bp[c];
#pragma unroll
    for (int r = 0; r < 8; ++r) {
      int t = 16 * mt + 8 * ((lane >> 4) & 1) + r;
      int rr = t >> 3, cc = t & 7;
      int h0 = (wr * WIN + rr + SHIFT) & 255;
      int w0 = (wc * WIN + cc + SHIFT) & 255;
      size_t addr = ((size_t)(b * 65536 + h0 * 256 + w0)) * CCH + c;
      xattn[addr] = x[addr] + acc[r] + bias;
    }
  }
}

// ---- K2: fused LN2 + W1 + GELU + depthwise3x3 + GELU + W2 + residual -------
// One 256-thread WG per 8x8 tile with 1-pixel halo; HID stays in LDS.
__global__ __launch_bounds__(256) void leff_kernel(
    const float* __restrict__ xattn, const float* __restrict__ ln2g,
    const float* __restrict__ ln2b, const float* __restrict__ b1f,
    const float* __restrict__ dwkf, const float* __restrict__ dwbf,
    const float* __restrict__ b2f, const bf16* __restrict__ w1T,
    const bf16* __restrict__ w2T, float* __restrict__ out) {
  extern __shared__ char smem[];
  bf16* xln  = (bf16*)smem;            // [112][64]  (100 halo tokens, padded)
  bf16* hidT = xln + 112 * 64;         // [256][112] channel-major hidden
  bf16* hid2 = hidT + 256 * 112;       // [64][256]  post-conv hidden
  int*  vld  = (int*)(hid2 + 64 * 256);// [112] halo validity

  const int tid = threadIdx.x, lane = tid & 31, wid = tid >> 5;
  const int blk = blockIdx.x;
  const int b = blk >> 10, ti = blk & 1023, ty = ti >> 5, tx = ti & 31;

  __builtin_prefetch(w1T, 0, 3);
  __builtin_prefetch(w2T, 0, 3);

  // LN over the 10x10 halo
  if (tid < 112) {
    int ok = 0;
    if (tid < 100) {
      int hr = tid / 10, hc = tid % 10;
      int gh = ty * 8 + hr - 1, gw = tx * 8 + hc - 1;
      ok = (gh >= 0 && gh < 256 && gw >= 0 && gw < 256);
      if (ok) {
        const float* px = xattn + ((size_t)(b * 65536 + gh * 256 + gw)) * CCH;
        float v[CCH]; float m = 0.f;
#pragma unroll
        for (int j = 0; j < CCH; ++j) { v[j] = px[j]; m += v[j]; }
        m *= (1.f / CCH);
        float var = 0.f;
#pragma unroll
        for (int j = 0; j < CCH; ++j) { float d = v[j] - m; var += d * d; }
        float rs = rsqrtf(var * (1.f / CCH) + 1e-5f);
#pragma unroll
        for (int j = 0; j < CCH; ++j)
          xln[tid * 64 + j] = (bf16)((v[j] - m) * rs * ln2g[j] + ln2b[j]);
      }
    }
    vld[tid] = ok;
    if (!ok)
      for (int j = 0; j < CCH; ++j) xln[tid * 64 + j] = (bf16)0.f;
  }
  __syncthreads();

  // GEMM1: [112x64] @ w1T -> GELU -> hidT (channel-major, invalid rows zeroed)
  for (int idx = wid; idx < 112; idx += 8) {
    int mt = idx >> 4, nt = idx & 15;
    f32x8 acc = {};
    for (int ks = 0; ks < 64; ks += 32) {
      bf16x16 a  = load_frag(xln, 16 * mt, ks, 64, lane);
      bf16x16 bm = load_frag(w1T, 16 * nt, ks, 64, lane);
      acc = WMMA_BF16(a, bm, acc);
    }
    int n = lane & 15;
    float bb = b1f[nt * 16 + n];
    int m0 = 16 * mt + 8 * ((lane >> 4) & 1);
    bf16x8 v;
#pragma unroll
    for (int r = 0; r < 8; ++r) {
      float g = gelu_exact(acc[r] + bb);
      v[r] = (bf16)(vld[m0 + r] ? g : 0.f);
    }
    *(bf16x8*)(hidT + (nt * 16 + n) * 112 + m0) = v;
  }
  __syncthreads();

  // depthwise 3x3: thread tid owns channel tid across the 8x8 interior
  {
    float k9[9];
#pragma unroll
    for (int j = 0; j < 9; ++j) k9[j] = dwkf[tid * 9 + j];
    float bc = dwbf[tid];
    const bf16* row = hidT + tid * 112;
    for (int t = 0; t < 64; ++t) {
      int r = t >> 3, c = t & 7;
      float acc = 0.f;
#pragma unroll
      for (int dy = 0; dy < 3; ++dy)
#pragma unroll
        for (int dx = 0; dx < 3; ++dx)
          acc += (float)row[(r + dy) * 10 + (c + dx)] * k9[dy * 3 + dx];
      hid2[t * 256 + tid] = (bf16)gelu_exact(acc + bc);
    }
  }
  __syncthreads();

  // GEMM2: [64x256] @ w2T -> +b2 +residual -> out
  for (int idx = wid; idx < 16; idx += 8) {
    int mt = idx >> 2, nt = idx & 3;
    f32x8 acc = {};
    for (int ks = 0; ks < 256; ks += 32) {
      bf16x16 a  = load_frag(hid2, 16 * mt, ks, 256, lane);
      bf16x16 bm = load_frag(w2T,  16 * nt, ks, 256, lane);
      acc = WMMA_BF16(a, bm, acc);
    }
    int n = lane & 15, c = nt * 16 + n;
    float bb = b2f[c];
#pragma unroll
    for (int r = 0; r < 8; ++r) {
      int t = 16 * mt + 8 * ((lane >> 4) & 1) + r;
      int gh = ty * 8 + (t >> 3), gw = tx * 8 + (t & 7);
      size_t addr = ((size_t)(b * 65536 + gh * 256 + gw)) * CCH + c;
      out[addr] = xattn[addr] + acc[r] + bb;
    }
  }
}

extern "C" void kernel_launch(void* const* d_in, const int* in_sizes, int n_in,
                              void* d_out, int out_size, void* d_ws, size_t ws_size,
                              hipStream_t stream) {
  const float* x      = (const float*)d_in[0];
  const float* ln1g   = (const float*)d_in[1];
  const float* ln1b   = (const float*)d_in[2];
  const float* rpb    = (const float*)d_in[3];
  const float* wq     = (const float*)d_in[4];
  const float* bq     = (const float*)d_in[5];
  const float* wkv    = (const float*)d_in[6];
  const float* bkv    = (const float*)d_in[7];
  const float* wp     = (const float*)d_in[8];
  const float* bp     = (const float*)d_in[9];
  const float* ln2g   = (const float*)d_in[10];
  const float* ln2b   = (const float*)d_in[11];
  const float* w1     = (const float*)d_in[12];
  const float* b1f    = (const float*)d_in[13];
  const float* dwkf   = (const float*)d_in[14];
  const float* dwbf   = (const float*)d_in[15];
  const float* w2     = (const float*)d_in[16];
  const float* b2f    = (const float*)d_in[17];
  (void)in_sizes; (void)n_in; (void)out_size; (void)ws_size;

  char*  ws     = (char*)d_ws;
  float* xattn  = (float*)ws;
  bf16*  wqkvT  = (bf16*)(ws + OFF_WQKVT);
  bf16*  wpT    = (bf16*)(ws + OFF_WPT);
  bf16*  w1T    = (bf16*)(ws + OFF_W1T);
  bf16*  w2T    = (bf16*)(ws + OFF_W2T);
  float* biasT  = (float*)(ws + OFF_BIAS);
  float* outp   = (float*)d_out;

  prep_kernel<<<256, 256, 0, stream>>>(wq, wkv, wp, w1, w2, rpb,
                                       wqkvT, wpT, w1T, w2T, biasT);

  // attn LDS: 81920 working set + 24576 + 8192 TDM-staged weights = 114688 B
  attn_kernel<<<8192, 128, 114688, stream>>>(x, ln1g, ln1b, bq, bkv, bp,
                                             wqkvT, wpT, biasT, xattn);

  // leff LDS: 14336+57344+32768+448 = 104896 B
  leff_kernel<<<8192, 256, 104896, stream>>>(xattn, ln2g, ln2b, b1f, dwkf, dwbf,
                                             b2f, w1T, w2T, outp);
}
